// myFeatureCombine_version1_6614249636394
// MI455X (gfx1250) — matrix operand order repaired and checked by reading, not compiled, not run on hardware
//
#include <hip/hip_runtime.h>
#include <hip/hip_fp16.h>

typedef __attribute__((ext_vector_type(16))) _Float16 v16h;
typedef __attribute__((ext_vector_type(8)))  float    v8f;

#define SCALE_ 0.1767766952966369f
#define XS 196   // row stride (halves) for x/q/k/attn rows of 192
#define VS 68    // row stride (halves) for vT/probs rows of 64

// LDS offsets in halves
#define OFF_X   0
#define OFF_Q   12544
#define OFF_K   25088
#define OFF_VT  37632
#define OFF_PR  50688
#define LDS_HALVES 59392   // 118784 bytes

// A-fragment K index for 16-bit A (16x32): vgpr v, lane-group g
__device__ __forceinline__ int a_kk(int v, int g) {
    return (g << 4) + (v << 1);
}
// B-fragment K index for 16-bit B (32x16): lane-group g holds K half
__device__ __forceinline__ int b_kk(int v, int g) {
    return (g << 4) + (v << 1);
}

union PkH2 { unsigned u; _Float16 h[2]; };
union PkH4 { uint2 u2; _Float16 h[4]; };

// ---------------- Kernel 0: pack weights into f16 B-fragment layout ----------------
__global__ __launch_bounds__(256)
void swin_pack_weights_kernel(const float* __restrict__ qkv_w,
                              const float* __restrict__ out_w,
                              _Float16* __restrict__ wqkv,
                              _Float16* __restrict__ wout)
{
    int i0 = blockIdx.x * blockDim.x + threadIdx.x;
    int stride = gridDim.x * blockDim.x;
    // qkv_w: 192 x 576 row-major (k, n) -> fragments [kstep*36 + ntile][lane*16 + v*2 + r]
    for (int idx = i0; idx < 192 * 576; idx += stride) {
        int k = idx / 576, n = idx - k * 576;
        int kt = k >> 5, kr = k & 31;
        int g = kr >> 4, rem = kr & 15, v = rem >> 1, r = rem & 1;
        int lanei = (g << 4) + (n & 15);
        int f = kt * 36 + (n >> 4);
        wqkv[f * 512 + lanei * 16 + v * 2 + r] = (_Float16)qkv_w[idx];
    }
    // out_w: 192 x 192 -> fragments [kstep*12 + ntile][...]
    for (int idx = i0; idx < 192 * 192; idx += stride) {
        int k = idx / 192, n = idx - k * 192;
        int kt = k >> 5, kr = k & 31;
        int g = kr >> 4, rem = kr & 15, v = rem >> 1, r = rem & 1;
        int lanei = (g << 4) + (n & 15);
        int f = kt * 12 + (n >> 4);
        wout[f * 512 + lanei * 16 + v * 2 + r] = (_Float16)out_w[idx];
    }
}

// ---------------- Fused windowed-attention kernel: 1 workgroup = 1 window ----------------
__global__ __launch_bounds__(256)
void swin_window_attn_kernel(const float* __restrict__ x,
                             const float* __restrict__ qkv_b,
                             const float* __restrict__ rel_pos,
                             const float* __restrict__ out_b,
                             const _Float16* __restrict__ wqkv,
                             const _Float16* __restrict__ wout,
                             float* __restrict__ out)
{
    extern __shared__ _Float16 smem[];
    _Float16* lds_x  = smem + OFF_X;    // 64 x XS ; aliased as attn-out later
    _Float16* lds_q  = smem + OFF_Q;    // 64 x XS
    _Float16* lds_k  = smem + OFF_K;    // 64 x XS
    _Float16* lds_vT = smem + OFF_VT;   // 192 x VS (channel-major)
    _Float16* lds_pr = smem + OFF_PR;   // 2 x 64 x VS

    const int wgid = blockIdx.x;            // 0..4095
    const int b    = wgid >> 10;
    const int win  = wgid & 1023;
    const int wi   = win >> 5, wj = win & 31;
    const int tid  = threadIdx.x;
    const int wave = tid >> 5, lane = tid & 31;
    const int lg   = lane >> 4, ln = lane & 15;
    const int mtile = wave & 3;
    const int grp   = wave >> 2;

    // ---- Phase 0: gather rolled x-window (64 tokens x 192 ch) -> LDS f16 ----
    // 64*192 floats = 3072 float4 quads; 256 threads x 12 iters, b128 loads.
    #pragma unroll
    for (int i = 0; i < 12; ++i) {
        int p  = tid + (i << 8);        // quad index, 0..3071
        int t  = p / 48;                // token 0..63
        int cq = p - t * 48;
        int ch = cq << 2;
        int pi = t >> 3, pj = t & 7;
        int h  = (wi * 8 + pi + 4) & 255;
        int w  = (wj * 8 + pj + 4) & 255;
        const float4 f = *(const float4*)(x + (((size_t)(b * 256 + h) * 256 + w) * 192 + ch));
        PkH4 pk;
        pk.h[0] = (_Float16)f.x; pk.h[1] = (_Float16)f.y;
        pk.h[2] = (_Float16)f.z; pk.h[3] = (_Float16)f.w;
        *(uint2*)(lds_x + t * XS + ch) = pk.u2;
    }
    __syncthreads();

    // ---- Phase 1: QKV = X(64x192) @ Wqkv(192x576) + b ----
    {
        v16h afr[6];
        const int m = mtile * 16 + ln;
        #pragma unroll
        for (int ks = 0; ks < 6; ++ks)
            #pragma unroll
            for (int v = 0; v < 8; ++v)
                ((unsigned*)&afr[ks])[v] =
                    *(const unsigned*)(lds_x + m * XS + ks * 32 + a_kk(v, lg));

        for (int j = 0; j < 18; ++j) {
            const int ntile = grp * 18 + j;
            const int col   = ntile * 16 + ln;
            const float bias = qkv_b[col];
            v8f acc = {bias, bias, bias, bias, bias, bias, bias, bias};
            #pragma unroll
            for (int ks = 0; ks < 6; ++ks) {
                v16h bf = *(const v16h*)(wqkv + ((size_t)(ks * 36 + ntile) * 512 + lane * 16));
                acc = __builtin_amdgcn_wmma_f32_16x16x32_f16(
                        false, afr[ks], false, bf, (short)0, acc, false, false);
            }
            if (ntile < 12) {                 // Q, row-major
                #pragma unroll
                for (int v = 0; v < 8; ++v) {
                    int t = mtile * 16 + v + (lg << 3);
                    lds_q[t * XS + col] = (_Float16)acc[v];
                }
            } else if (ntile < 24) {          // K, row-major
                int ch = col - 192;
                #pragma unroll
                for (int v = 0; v < 8; ++v) {
                    int t = mtile * 16 + v + (lg << 3);
                    lds_k[t * XS + ch] = (_Float16)acc[v];
                }
            } else {                          // V, transposed (channel-major)
                int ch = col - 384;
                #pragma unroll
                for (int v = 0; v < 8; v += 2) {
                    int t = mtile * 16 + v + (lg << 3);
                    PkH2 pk; pk.h[0] = (_Float16)acc[v]; pk.h[1] = (_Float16)acc[v + 1];
                    *(unsigned*)(lds_vT + ch * VS + t) = pk.u;
                }
            }
        }
    }
    __syncthreads();

    // ---- Phase 2: attention; 2 heads in parallel (grp), wave = 16 queries ----
    const int slot = grp;
    for (int hp = 0; hp < 3; ++hp) {
        const int head = hp * 2 + grp;
        // A-fragment of Q (16 queries x 32 ch)
        v16h qa;
        {
            int m = mtile * 16 + ln;
            #pragma unroll
            for (int v = 0; v < 8; ++v)
                ((unsigned*)&qa)[v] =
                    *(const unsigned*)(lds_q + m * XS + head * 32 + a_kk(v, lg));
        }
        v8f s[4];
        #pragma unroll
        for (int nt = 0; nt < 4; ++nt) {
            int key = nt * 16 + ln;
            v16h kb;
            #pragma unroll
            for (int v = 0; v < 8; ++v)
                ((unsigned*)&kb)[v] =
                    *(const unsigned*)(lds_k + key * XS + head * 32 + b_kk(v, lg));
            v8f z = {0.f, 0.f, 0.f, 0.f, 0.f, 0.f, 0.f, 0.f};
            s[nt] = __builtin_amdgcn_wmma_f32_16x16x32_f16(
                        false, qa, false, kb, (short)0, z, false, false);
            // epilogue: scale + relative-position bias + shift mask
            const int qi = key >> 3, qj = key & 7;
            #pragma unroll
            for (int v = 0; v < 8; ++v) {
                int p_ = mtile * 16 + v + (lg << 3);
                int pi = p_ >> 3, pj = p_ & 7;
                float val = s[nt][v] * SCALE_ +
                            rel_pos[head * 225 + (pi - qi + 7) * 15 + (pj - qj + 7)];
                bool msk = ((wi == 31) && ((pi < 4) != (qi < 4))) ||
                           ((wj == 31) && ((pj < 4) != (qj < 4)));
                s[nt][v] = msk ? -3.0e38f : val;
            }
        }
        // register softmax over 64 keys (rows live in one 16-lane group)
        #pragma unroll
        for (int v = 0; v < 8; ++v) {
            float m0 = fmaxf(fmaxf(s[0][v], s[1][v]), fmaxf(s[2][v], s[3][v]));
            #pragma unroll
            for (int o = 8; o >= 1; o >>= 1) m0 = fmaxf(m0, __shfl_xor(m0, o, 16));
            float sum = 0.f;
            #pragma unroll
            for (int nt = 0; nt < 4; ++nt) {
                float e = __expf(s[nt][v] - m0);
                s[nt][v] = e; sum += e;
            }
            #pragma unroll
            for (int o = 8; o >= 1; o >>= 1) sum += __shfl_xor(sum, o, 16);
            float inv = 1.0f / sum;
            #pragma unroll
            for (int nt = 0; nt < 4; ++nt) s[nt][v] *= inv;
        }
        // probs -> LDS f16 (row-major [query][key])
        #pragma unroll
        for (int nt = 0; nt < 4; ++nt) {
            int q_ = nt * 16 + ln;
            #pragma unroll
            for (int v = 0; v < 8; ++v) {
                int p_ = mtile * 16 + v + (lg << 3);
                lds_pr[slot * 4352 + p_ * VS + q_] = (_Float16)s[nt][v];
            }
        }
        __syncthreads();

        // ---- PV: out(16q x 32ch) = probs(16x64) @ V(64x32) ----
        v16h pa[2];
        {
            int m = mtile * 16 + ln;
            #pragma unroll
            for (int ks = 0; ks < 2; ++ks)
                #pragma unroll
                for (int v = 0; v < 8; ++v)
                    ((unsigned*)&pa[ks])[v] =
                        *(const unsigned*)(lds_pr + slot * 4352 + m * VS + ks * 32 + a_kk(v, lg));
        }
        #pragma unroll
        for (int nt = 0; nt < 2; ++nt) {
            int ch = head * 32 + nt * 16 + ln;
            v8f acc = {0.f, 0.f, 0.f, 0.f, 0.f, 0.f, 0.f, 0.f};
            #pragma unroll
            for (int ks = 0; ks < 2; ++ks) {
                v16h vb;
                #pragma unroll
                for (int v = 0; v < 8; ++v)
                    ((unsigned*)&vb)[v] =
                        *(const unsigned*)(lds_vT + ch * VS + ks * 32 + b_kk(v, lg));
                acc = __builtin_amdgcn_wmma_f32_16x16x32_f16(
                        false, pa[ks], false, vb, (short)0, acc, false, false);
            }
            #pragma unroll
            for (int v = 0; v < 8; ++v) {
                int t = mtile * 16 + v + (lg << 3);
                lds_x[t * XS + ch] = (_Float16)acc[v];   // attn-out aliases lds_x
            }
        }
        __syncthreads();
    }

    // ---- Phase 3: out-proj (64x192 @ 192x192) + bias, scatter with un-shift ----
    {
        v16h aof[6];
        const int m = mtile * 16 + ln;
        #pragma unroll
        for (int ks = 0; ks < 6; ++ks)
            #pragma unroll
            for (int v = 0; v < 8; ++v)
                ((unsigned*)&aof[ks])[v] =
                    *(const unsigned*)(lds_x + m * XS + ks * 32 + a_kk(v, lg));

        size_t rowbase[8];
        #pragma unroll
        for (int v = 0; v < 8; ++v) {
            int t  = mtile * 16 + v + (lg << 3);
            int pi = t >> 3, pj = t & 7;
            int h  = (wi * 8 + pi + 4) & 255;
            int w  = (wj * 8 + pj + 4) & 255;
            rowbase[v] = ((size_t)(b * 256 + h) * 256 + w) * 192;
        }
        #pragma unroll
        for (int j = 0; j < 6; ++j) {
            const int ntile = grp * 6 + j;
            const int col   = ntile * 16 + ln;
            const float bias = out_b[col];
            v8f acc = {bias, bias, bias, bias, bias, bias, bias, bias};
            #pragma unroll
            for (int ks = 0; ks < 6; ++ks) {
                v16h bf = *(const v16h*)(wout + ((size_t)(ks * 12 + ntile) * 512 + lane * 16));
                acc = __builtin_amdgcn_wmma_f32_16x16x32_f16(
                        false, aof[ks], false, bf, (short)0, acc, false, false);
            }
            #pragma unroll
            for (int v = 0; v < 8; ++v)
                out[rowbase[v] + col] = acc[v];
        }
    }
}

extern "C" void kernel_launch(void* const* d_in, const int* in_sizes, int n_in,
                              void* d_out, int out_size, void* d_ws, size_t ws_size,
                              hipStream_t stream)
{
    const float* x       = (const float*)d_in[0];
    const float* qkv_w   = (const float*)d_in[1];
    const float* qkv_b   = (const float*)d_in[2];
    const float* rel_pos = (const float*)d_in[3];
    const float* out_w   = (const float*)d_in[4];
    const float* out_b   = (const float*)d_in[5];

    _Float16* wqkv = (_Float16*)d_ws;            // 192*576 halves
    _Float16* wout = wqkv + 192 * 576;           // 192*192 halves

    swin_pack_weights_kernel<<<256, 256, 0, stream>>>(qkv_w, out_w, wqkv, wout);

    const int shmem = LDS_HALVES * (int)sizeof(_Float16);   // 118784 B
    (void)hipFuncSetAttribute((const void*)swin_window_attn_kernel,
                              hipFuncAttributeMaxDynamicSharedMemorySize, shmem);
    swin_window_attn_kernel<<<4096, 256, shmem, stream>>>(
        x, qkv_b, rel_pos, out_b, wqkv, wout, (float*)d_out);
}
